// GRUDCell_90941637525969
// MI455X (gfx1250) — compile-verified, matrix operand order checked by reading, not run on hardware
//
#include <hip/hip_runtime.h>
#include <hip/hip_bf16.h>
#include <math.h>

// ---------------------------------------------------------------------------
// GRU-D cell, fused, f16 WMMA (f32 accumulate) for gfx1250 (MI455X).
// B=8192, IN=256, HID=1024.
// Kernel 3 uses double-buffered GLOBAL_LOAD_ASYNC_TO_LDS_B128 staging
// (ASYNCcnt) + ds_load_b128 operand fetch + v_wmma_f32_16x16x32_f16.
// ---------------------------------------------------------------------------

typedef _Float16 h16;
typedef __attribute__((ext_vector_type(16))) _Float16 v16h;
typedef __attribute__((ext_vector_type(8)))  _Float16 v8h;
typedef __attribute__((ext_vector_type(8)))  float    v8f;

#define B_DIM   8192
#define IN_DIM  256
#define HID_DIM 1024
#define KCAT    1536   // 2*IN + HID (concatenated K for r/z gates)
#define N3H     3072   // 3*HID

// LDS tiling for kernel 3
#define BM      128          // block rows
#define BN      32           // block hidden cols
#define BK      64           // K slab per pipeline stage
#define LDW     72           // padded LDS row length in halves (144B, 16B mult)
#define A_ROWS  BM           // 128
#define W_ROWS  96           // 32 cols x {r,z,n}
#define A_HALVES (A_ROWS * LDW)   // per buffer
#define W_HALVES (W_ROWS * LDW)

union Frag16 { v16h v; v8h h[2]; };

// 16-bit A-operand layout (B is its dual with lane = N over weight rows):
// lanes 0-15 carry K = kb+{0..7, 16..23}, lanes 16-31 K = kb+{8..15, 24..31}.
__device__ __forceinline__ v16h load_frag(const h16* __restrict__ p, int kb, int laneHi) {
    const h16* q = p + kb + laneHi * 8;
    Frag16 f;
    f.h[0] = *reinterpret_cast<const v8h*>(q);
    f.h[1] = *reinterpret_cast<const v8h*>(q + 16);
    return f.v;
}

__device__ __forceinline__ v8f wmma_f16(v16h a, v16h b, v8f c) {
    return __builtin_amdgcn_wmma_f32_16x16x32_f16(false, a, false, b, (short)0, c, false, false);
}

__device__ __forceinline__ float sigmoidf_(float x) {
    return 1.0f / (1.0f + __expf(-x));
}

// Async global->LDS 16B copy (per-lane address), tracked by ASYNCcnt.
__device__ __forceinline__ void async_g2l_b128(unsigned lds_byte_off, const void* gaddr) {
    asm volatile("global_load_async_to_lds_b128 %0, %1, off"
                 :: "v"(lds_byte_off), "v"((unsigned long long)gaddr)
                 : "memory");
}

// ---------------------------------------------------------------------------
// Kernel 1: pack f32 inputs/weights into f16 workspace layouts.
// ---------------------------------------------------------------------------
__global__ void __launch_bounds__(256) grud_prep_kernel(
    const float* __restrict__ delta, const float* __restrict__ m,
    const float* __restrict__ Wgx,   const float* __restrict__ Wgh,
    const float* __restrict__ W_ih,  const float* __restrict__ W_hh,
    h16* __restrict__ dlt16, h16* __restrict__ wg16,
    h16* __restrict__ acat,  h16* __restrict__ wcat)
{
    const long Nd  = (long)B_DIM * IN_DIM;
    const long Nm  = (long)B_DIM * IN_DIM;
    const long Nwg = (long)(IN_DIM + HID_DIM) * IN_DIM;
    const long Nwc = (long)N3H * KCAT;
    const long total = Nd + Nm + Nwg + Nwc;
    for (long i = (long)blockIdx.x * blockDim.x + threadIdx.x; i < total;
         i += (long)gridDim.x * blockDim.x) {
        if (i < Nd) {
            dlt16[i] = (h16)delta[i];
        } else if (i < Nd + Nm) {
            long e = i - Nd;
            long r = e / IN_DIM, c = e - r * IN_DIM;
            acat[r * KCAT + IN_DIM + c] = (h16)m[e];
        } else if (i < Nd + Nm + Nwg) {
            long e = i - Nd - Nm;
            float v = (e < (long)IN_DIM * IN_DIM) ? Wgx[e] : Wgh[e - (long)IN_DIM * IN_DIM];
            wg16[e] = (h16)v;
        } else {
            long e = i - Nd - Nm - Nwg;
            long n = e / KCAT, k = e - n * KCAT;
            float v = (k < 2 * IN_DIM) ? W_ih[n * (2 * IN_DIM) + k]
                                       : W_hh[n * HID_DIM + (k - 2 * IN_DIM)];
            wcat[e] = (h16)v;
        }
    }
}

// ---------------------------------------------------------------------------
// Kernel 2: gamma GEMM (M=8192, K=256, N=1280) + decay/imputation epilogue.
// ---------------------------------------------------------------------------
__global__ void __launch_bounds__(256) grud_gamma_kernel(
    const h16* __restrict__ dlt16, const h16* __restrict__ wg16,
    const float* __restrict__ x,      const float* __restrict__ m,
    const float* __restrict__ x_prev, const float* __restrict__ x_mean,
    const float* __restrict__ h_prev,
    const float* __restrict__ bgx,    const float* __restrict__ bgh,
    float* __restrict__ out_xt, h16* __restrict__ acat, float* __restrict__ ht32)
{
    const int lane = threadIdx.x & 31;
    const int wave = threadIdx.x >> 5;
    const int lm = lane & 15, lh = lane >> 4;
    const int m0 = blockIdx.x * 128 + (wave & 3) * 32;
    const int nb = blockIdx.y * 32 + (wave >> 2) * 16;
    const int ncol = nb + lm;

    const h16* __restrict__ wrow  = wg16  + (size_t)ncol * IN_DIM;
    const h16* __restrict__ arow0 = dlt16 + (size_t)(m0 + lm) * IN_DIM;
    const h16* __restrict__ arow1 = dlt16 + (size_t)(m0 + 16 + lm) * IN_DIM;

    v8f acc0 = {}; v8f acc1 = {};
#pragma unroll
    for (int kb = 0; kb < IN_DIM; kb += 32) {
        v16h b  = load_frag(wrow,  kb, lh);
        v16h a0 = load_frag(arow0, kb, lh);
        v16h a1 = load_frag(arow1, kb, lh);
        acc0 = wmma_f16(a0, b, acc0);
        acc1 = wmma_f16(a1, b, acc1);
    }

#pragma unroll
    for (int s = 0; s < 2; ++s) {
        v8f acc = s ? acc1 : acc0;
        const int mb = m0 + s * 16 + lh * 8;
#pragma unroll
        for (int vi = 0; vi < 8; ++vi) {
            const int row = mb + vi;
            if (ncol < IN_DIM) {
                const float g = __expf(-fmaxf(acc[vi] + bgx[ncol], 0.0f));
                const size_t idx = (size_t)row * IN_DIM + ncol;
                const float mv = m[idx];
                const float xt = mv * x[idx] +
                                 (1.0f - mv) * (g * x_prev[idx] + (1.0f - g) * x_mean[ncol]);
                out_xt[idx] = xt;
                acat[(size_t)row * KCAT + ncol] = (h16)xt;
            } else {
                const int hc = ncol - IN_DIM;
                const float g = __expf(-fmaxf(acc[vi] + bgh[hc], 0.0f));
                const size_t idx = (size_t)row * HID_DIM + hc;
                const float ht = g * h_prev[idx];
                ht32[idx] = ht;
                acat[(size_t)row * KCAT + 2 * IN_DIM + hc] = (h16)ht;
            }
        }
    }
}

// ---------------------------------------------------------------------------
// Kernel 3: fused GRU GEMM + gate epilogue with async-LDS double buffering.
// A = A_cat (B x 1536 f16), W = W_cat (3072 x 1536 f16).
// Block tile 128x32, K slab 64, 2 LDS buffers.
// Per slab each thread issues 7 GLOBAL_LOAD_ASYNC_TO_LDS_B128 (A: 4, W: 3);
// s_wait_asynccnt 7 retires the previous slab while the next is in flight.
// ---------------------------------------------------------------------------
__global__ void __launch_bounds__(256) grud_gru_kernel(
    const h16* __restrict__ acat, const h16* __restrict__ wcat,
    const float* __restrict__ b_ih, const float* __restrict__ b_hh,
    const float* __restrict__ ht32, float* __restrict__ h_new)
{
    __shared__ __align__(16) h16 smem[2 * A_HALVES + 2 * W_HALVES]; // 63 KB

    const int t    = threadIdx.x;
    const int lane = t & 31;
    const int wave = t >> 5;
    const int lm = lane & 15, lh = lane >> 4;
    const int m0 = blockIdx.x * BM + (wave & 3) * 32;   // global row base of this wave
    const int nb0 = blockIdx.y * BN;                    // block hidden-col base
    const int ncol = nb0 + (wave >> 2) * 16 + lm;       // this lane's hidden col

    const h16* __restrict__ acat_blk = acat + (size_t)blockIdx.x * BM * KCAT;
    const unsigned smemBase = (unsigned)(size_t)(void*)smem; // LDS byte offset
    const unsigned asBase[2] = { smemBase,
                                 smemBase + A_HALVES * (unsigned)sizeof(h16) };
    const unsigned wsBase[2] = { smemBase + 2u * A_HALVES * (unsigned)sizeof(h16),
                                 smemBase + (2u * A_HALVES + W_HALVES) * (unsigned)sizeof(h16) };

    // Issue one K-slab of async copies into buffer `buf` for K window [kb, kb+64).
    auto issue_slab = [&](int buf, int kb) {
        // A: 128 rows x 64 halves = 1024 16B chunks, 4 per thread
#pragma unroll
        for (int i = 0; i < 4; ++i) {
            const int cid = t + i * 256;
            const int row = cid >> 3, c = cid & 7;
            async_g2l_b128(asBase[buf] + (unsigned)(row * (LDW * 2) + c * 16),
                           acat_blk + (size_t)row * KCAT + kb + c * 8);
        }
        // W: 96 rows (r:0-31, z:32-63, n:64-95) x 64 halves = 768 chunks, 3 per thread
#pragma unroll
        for (int i = 0; i < 3; ++i) {
            const int cid = t + i * 256;
            const int row = cid >> 3, c = cid & 7;
            const int grow = (row < 32) ? (nb0 + row)
                           : (row < 64) ? (HID_DIM + nb0 + (row - 32))
                                        : (2 * HID_DIM + nb0 + (row - 64));
            async_g2l_b128(wsBase[buf] + (unsigned)(row * (LDW * 2) + c * 16),
                           wcat + (size_t)grow * KCAT + kb + c * 8);
        }
    };

    v8f ar0 = {}, ar1 = {}, az0 = {}, az1 = {};
    v8f ani0 = {}, ani1 = {}, anh0 = {}, anh1 = {};

    const int NSTEPS = KCAT / BK;       // 24 slabs; slabs 0..7 feed i_n, 8..23 feed h_n
    issue_slab(0, 0);

    for (int s = 0; s < NSTEPS; ++s) {
        const int buf = s & 1;
        if (s + 1 < NSTEPS) {
            issue_slab(buf ^ 1, (s + 1) * BK);
            asm volatile("s_wait_asynccnt 0x7" ::: "memory"); // previous slab landed
        } else {
            asm volatile("s_wait_asynccnt 0x0" ::: "memory");
        }
        __syncthreads();

        const h16* aBuf = smem + buf * A_HALVES;
        const h16* wBuf = smem + 2 * A_HALVES + buf * W_HALVES;
        const h16* ar0p = aBuf + (size_t)((wave & 3) * 32 + lm) * LDW;
        const h16* ar1p = ar0p + 16 * LDW;
        const h16* wrp  = wBuf + (size_t)((wave >> 2) * 16 + lm) * LDW;
        const h16* wzp  = wrp + 32 * LDW;
        const h16* wnp  = wrp + 64 * LDW;

#pragma unroll
        for (int kk = 0; kk < BK; kk += 32) {
            v16h a0 = load_frag(ar0p, kk, lh);
            v16h a1 = load_frag(ar1p, kk, lh);
            v16h br = load_frag(wrp,  kk, lh);
            v16h bz = load_frag(wzp,  kk, lh);
            v16h bn = load_frag(wnp,  kk, lh);
            ar0 = wmma_f16(a0, br, ar0);  ar1 = wmma_f16(a1, br, ar1);
            az0 = wmma_f16(a0, bz, az0);  az1 = wmma_f16(a1, bz, az1);
            if (s < (2 * IN_DIM) / BK) {            // K < 512: gru_in part -> i_n
                ani0 = wmma_f16(a0, bn, ani0);  ani1 = wmma_f16(a1, bn, ani1);
            } else {                                // K >= 512: h_tilde part -> h_n
                anh0 = wmma_f16(a0, bn, anh0);  anh1 = wmma_f16(a1, bn, anh1);
            }
        }
        __syncthreads();   // protect buffer from next slab's async writes
    }

    const float br_ = b_ih[ncol]               + b_hh[ncol];
    const float bz_ = b_ih[ncol + HID_DIM]     + b_hh[ncol + HID_DIM];
    const float bin = b_ih[ncol + 2 * HID_DIM];
    const float bhn = b_hh[ncol + 2 * HID_DIM];

#pragma unroll
    for (int sI = 0; sI < 2; ++sI) {
        v8f ar  = sI ? ar1  : ar0;
        v8f az  = sI ? az1  : az0;
        v8f ani = sI ? ani1 : ani0;
        v8f anh = sI ? anh1 : anh0;
        const int mb = m0 + sI * 16 + lh * 8;
#pragma unroll
        for (int vi = 0; vi < 8; ++vi) {
            const int row = mb + vi;
            const float r = sigmoidf_(ar[vi] + br_);
            const float z = sigmoidf_(az[vi] + bz_);
            const float n = tanhf(ani[vi] + bin + r * (anh[vi] + bhn));
            const size_t idx = (size_t)row * HID_DIM + ncol;
            const float ht = ht32[idx];
            h_new[idx] = (1.0f - z) * n + z * ht;
        }
    }
}

// ---------------------------------------------------------------------------
extern "C" void kernel_launch(void* const* d_in, const int* in_sizes, int n_in,
                              void* d_out, int out_size, void* d_ws, size_t ws_size,
                              hipStream_t stream) {
    (void)in_sizes; (void)n_in; (void)out_size; (void)ws_size;

    const float* x      = (const float*)d_in[0];
    const float* m      = (const float*)d_in[1];
    const float* delta  = (const float*)d_in[2];
    const float* h_prev = (const float*)d_in[3];
    const float* x_prev = (const float*)d_in[4];
    const float* x_mean = (const float*)d_in[5];
    const float* Wgx    = (const float*)d_in[6];
    const float* bgx    = (const float*)d_in[7];
    const float* Wgh    = (const float*)d_in[8];
    const float* bgh    = (const float*)d_in[9];
    const float* W_ih   = (const float*)d_in[10];
    const float* b_ih   = (const float*)d_in[11];
    const float* W_hh   = (const float*)d_in[12];
    const float* b_hh   = (const float*)d_in[13];

    float* h_new  = (float*)d_out;                    // (B, HID)
    float* out_xt = h_new + (size_t)B_DIM * HID_DIM;  // (B, IN)

    char* ws = (char*)d_ws;
    size_t off = 0;
    auto carve = [&](size_t bytes) -> char* {
        char* p = ws + off;
        off = (off + bytes + 255) & ~(size_t)255;
        return p;
    };
    h16*   dlt16 = (h16*)  carve((size_t)B_DIM * IN_DIM * sizeof(h16));
    h16*   wg16  = (h16*)  carve((size_t)(IN_DIM + HID_DIM) * IN_DIM * sizeof(h16));
    h16*   acat  = (h16*)  carve((size_t)B_DIM * KCAT * sizeof(h16));
    h16*   wcat  = (h16*)  carve((size_t)N3H * KCAT * sizeof(h16));
    float* ht32  = (float*)carve((size_t)B_DIM * HID_DIM * sizeof(float));

    grud_prep_kernel<<<2048, 256, 0, stream>>>(delta, m, Wgx, Wgh, W_ih, W_hh,
                                               dlt16, wg16, acat, wcat);
    grud_gamma_kernel<<<dim3(64, 40), 256, 0, stream>>>(
        dlt16, wg16, x, m, x_prev, x_mean, h_prev, bgx, bgh, out_xt, acat, ht32);
    grud_gru_kernel<<<dim3(64, 32), 256, 0, stream>>>(
        acat, wcat, b_ih, b_hh, ht32, h_new);
}
